// GPT_59382217834930
// MI455X (gfx1250) — matrix operand order repaired and checked by reading, not compiled, not run on hardware
//
#include <hip/hip_runtime.h>
#include <math.h>

// GPT-2 small forward for gfx1250 (MI455X). All GEMMs use v_wmma_f32_16x16x32_bf16
// (fp32->bf16 staged through LDS, fp32 accumulation). Attention is a flash-style
// wave-per-16-query kernel with WMMA for QK^T and P*V.

#define Bb 4
#define Tt 1024
#define LL 12
#define HH 12
#define CC 768
#define DD 64
#define FFF 3072
#define VV 16384
#define MT (Bb * Tt)

#define NEGINF (-__builtin_inff())

typedef __attribute__((ext_vector_type(16))) __bf16 v16bf;
typedef __attribute__((ext_vector_type(8))) float v8f;

union ABFrag { v16bf v; unsigned u[8]; };

__device__ inline unsigned short bf16_bits(float f) {
  union { __bf16 b; unsigned short u; } c; c.b = (__bf16)f; return c.u;
}
__device__ inline unsigned pk2(float lo, float hi) {
  return (unsigned)bf16_bits(lo) | ((unsigned)bf16_bits(hi) << 16);
}
__device__ inline float half_max(float v) {
#pragma unroll
  for (int m = 1; m < 16; m <<= 1) v = fmaxf(v, __shfl_xor(v, m, 16));
  return v;
}
__device__ inline float half_sum(float v) {
#pragma unroll
  for (int m = 1; m < 16; m <<= 1) v += __shfl_xor(v, m, 16);
  return v;
}

// ---------------------------------------------------------------- embedding
__global__ __launch_bounds__(256) void embed_kernel(
    const int* __restrict__ idx, const float* __restrict__ tok,
    const float* __restrict__ pos, float* __restrict__ x) {
  int i = blockIdx.x * 256 + threadIdx.x;   // over MT*CC, exact multiple
  int c = i % CC;
  int mt = i / CC;
  int t = mt % Tt;
  x[i] = tok[(size_t)idx[mt] * CC + c] + pos[t * CC + c];
}

// ---------------------------------------------------------------- layernorm
__global__ __launch_bounds__(256) void ln_kernel(
    const float* __restrict__ x, const float* __restrict__ w,
    const float* __restrict__ b, float* __restrict__ o, int n) {
  __shared__ float red[256];
  const int row = blockIdx.x;
  const float* xr = x + (size_t)row * n;
  float s = 0.f;
  for (int i = threadIdx.x; i < n; i += 256) s += xr[i];
  red[threadIdx.x] = s;
  __syncthreads();
  for (int off = 128; off > 0; off >>= 1) {
    if ((int)threadIdx.x < off) red[threadIdx.x] += red[threadIdx.x + off];
    __syncthreads();
  }
  const float mean = red[0] / n;
  __syncthreads();
  float vs = 0.f;
  for (int i = threadIdx.x; i < n; i += 256) { float d = xr[i] - mean; vs += d * d; }
  red[threadIdx.x] = vs;
  __syncthreads();
  for (int off = 128; off > 0; off >>= 1) {
    if ((int)threadIdx.x < off) red[threadIdx.x] += red[threadIdx.x + off];
    __syncthreads();
  }
  const float rstd = rsqrtf(red[0] / n + 1e-5f);
  float* orow = o + (size_t)row * n;
  for (int i = threadIdx.x; i < n; i += 256)
    orow[i] = (xr[i] - mean) * rstd * w[i] + b[i];
}

// ------------------------------------------------------- GEMM: out = A @ W^T (+bias,+act,+resid)
// A:[M,K] f32 row-major, W:[N,K] f32 row-major (nn.Linear weight), out:[M,N] f32.
// Block: 256 threads (8 waves), tile 64(M) x 128(N), K step 32 (bf16 WMMA K).
__global__ __launch_bounds__(256) void gemm_kernel(
    const float* __restrict__ A, const float* __restrict__ W,
    const float* __restrict__ bias, const float* __restrict__ resid,
    float* __restrict__ out, int M, int N, int K, int act) {
  __shared__ unsigned sA[64 * 16];    // 64 rows x 32 bf16 (16 u32)
  __shared__ unsigned sW[128 * 16];   // 128 rows x 32 bf16

  const int tid = threadIdx.x;
  const int lane = tid & 31;
  const int wave = tid >> 5;
  const int mi = wave & 3;        // M-tile 0..3
  const int nset = wave >> 2;     // covers N-tiles nset*4 .. nset*4+3
  const int m0 = blockIdx.y * 64;
  const int n0 = blockIdx.x * 128;

  v8f acc[4] = {};

  for (int k0 = 0; k0 < K; k0 += 32) {
    // stage A tile: thread -> row tid>>2, 8 floats at col 8*(tid&3)
    {
      int r = tid >> 2, c8 = (tid & 3) * 8;
      const float* src = A + (size_t)(m0 + r) * K + k0 + c8;
      unsigned* dst = sA + r * 16 + (c8 >> 1);
#pragma unroll
      for (int j = 0; j < 4; ++j) dst[j] = pk2(src[2 * j], src[2 * j + 1]);
    }
    // stage W tile: thread -> row tid>>1, 16 floats at col 16*(tid&1)
    {
      int r = tid >> 1, c16 = (tid & 1) * 16;
      const float* src = W + (size_t)(n0 + r) * K + k0 + c16;
      unsigned* dst = sW + r * 16 + (c16 >> 1);
#pragma unroll
      for (int j = 0; j < 8; ++j) dst[j] = pk2(src[2 * j], src[2 * j + 1]);
    }
    if (k0 + 32 < K) {
      __builtin_prefetch(A + (size_t)(m0 + (tid >> 2)) * K + k0 + 32, 0, 3);
      __builtin_prefetch(W + (size_t)(n0 + (tid >> 1)) * K + k0 + 32, 0, 3);
    }
    __syncthreads();

    // A fragment (contiguous K: lanes<16 K0..15, lanes>=16 K16..31)
    ABFrag a;
    {
      const unsigned* pa = sA + (mi * 16 + (lane & 15)) * 16;
      const int b4 = (lane & 16) ? 8 : 0;
#pragma unroll
      for (int j = 0; j < 4; ++j) { a.u[j] = pa[b4 + j]; a.u[4 + j] = pa[4 + b4 + j]; }
    }
#pragma unroll
    for (int t = 0; t < 4; ++t) {
      // B fragment (32x16): lanes<16 K0..15, lanes>=16 K16..31, col = W row
      ABFrag bf;
      const unsigned* pb = sW + ((nset * 4 + t) * 16 + (lane & 15)) * 16 + ((lane & 16) ? 8 : 0);
#pragma unroll
      for (int j = 0; j < 8; ++j) bf.u[j] = pb[j];
      acc[t] = __builtin_amdgcn_wmma_f32_16x16x32_bf16(
          false, a.v, false, bf.v, (short)0, acc[t], false, false);
    }
    __syncthreads();
  }

  // epilogue: C layout — lane<16: (M=r, N=lane); lane>=16: (M=8+r, N=lane-16)
  const int hi8 = (lane & 16) ? 8 : 0;
#pragma unroll
  for (int t = 0; t < 4; ++t) {
    const int nc = n0 + (nset * 4 + t) * 16 + (lane & 15);
    const float bvv = bias ? bias[nc] : 0.f;
#pragma unroll
    for (int r = 0; r < 8; ++r) {
      const int mr = m0 + mi * 16 + r + hi8;
      float v = acc[t][r] + bvv;
      if (act == 1) v = 0.5f * v * (1.f + erff(v * 0.70710678118654752f));
      if (resid) v += resid[(size_t)mr * N + nc];
      out[(size_t)mr * N + nc] = v;
    }
  }
}

// ------------------------------------------------------- flash attention
// grid: (Tt/64, HH, Bb), block: 128 threads (4 waves, 16 queries each).
__global__ __launch_bounds__(128) void attn_kernel(
    const float* __restrict__ qp, const float* __restrict__ kp,
    const float* __restrict__ vp, float* __restrict__ yp) {
  __shared__ unsigned sK[32 * 32];            // 32 keys x 64 dims bf16
  __shared__ unsigned short sVh[64 * 32];     // V^T: 64 dims x 32 keys bf16
  __shared__ unsigned sQ[4][16 * 32];         // per-wave Q 16x64 bf16
  __shared__ unsigned short sP[4][16 * 32];   // per-wave P 16x32 bf16

  const int tid = threadIdx.x;
  const int lane = tid & 31;
  const int w = tid >> 5;
  const int bz = blockIdx.z, hh = blockIdx.y;
  const int q0b = blockIdx.x * 64;
  const int q0 = q0b + w * 16;
  const int hi8 = (lane & 16) ? 8 : 0;
  const int n = lane & 15;

  // stage this wave's Q tile as bf16
  {
    int row = lane >> 1;
    int dbase = (lane & 1) * 32;
    const float* qs = qp + (size_t)(bz * Tt + q0 + row) * CC + hh * DD + dbase;
    unsigned* dst = sQ[w] + row * 32 + (dbase >> 1);
#pragma unroll
    for (int j = 0; j < 16; ++j) dst[j] = pk2(qs[2 * j], qs[2 * j + 1]);
  }
  asm volatile("s_wait_dscnt 0" ::: "memory");
  ABFrag aq[2];
#pragma unroll
  for (int s = 0; s < 2; ++s) {
    const unsigned* pq = sQ[w] + (lane & 15) * 32 + s * 16;
    const int b4 = (lane & 16) ? 8 : 0;
#pragma unroll
    for (int j = 0; j < 4; ++j) { aq[s].u[j] = pq[b4 + j]; aq[s].u[4 + j] = pq[4 + b4 + j]; }
  }

  float mstate[8], lstate[8];
#pragma unroll
  for (int r = 0; r < 8; ++r) { mstate[r] = NEGINF; lstate[r] = 0.f; }
  v8f o[4] = {};

  for (int j0 = 0; j0 < q0b + 64; j0 += 32) {
    // cooperative stage K (row-major) and V^T
    {
      int key = tid >> 2;
      int dbase = (tid & 3) * 16;
      const size_t base = (size_t)(bz * Tt + j0 + key) * CC + hh * DD + dbase;
      const float* ks = kp + base;
      unsigned* dk = sK + key * 32 + (dbase >> 1);
#pragma unroll
      for (int j = 0; j < 8; ++j) dk[j] = pk2(ks[2 * j], ks[2 * j + 1]);
      const float* vs = vp + base;
#pragma unroll
      for (int j = 0; j < 16; ++j) sVh[(dbase + j) * 32 + key] = bf16_bits(vs[j]);
    }
    __syncthreads();

    if (j0 <= q0 + 15) {
      // S = Q @ K^T (two 16x16 key tiles, K-dim 64 = 2 WMMA steps)
      v8f s0 = {}, s1 = {};
#pragma unroll
      for (int s = 0; s < 2; ++s) {
        ABFrag bk0, bk1;
        const int b8 = (lane & 16) ? 8 : 0;
        const unsigned* p0 = sK + (lane & 15) * 32 + s * 16 + b8;
        const unsigned* p1 = sK + (16 + (lane & 15)) * 32 + s * 16 + b8;
#pragma unroll
        for (int j = 0; j < 8; ++j) { bk0.u[j] = p0[j]; bk1.u[j] = p1[j]; }
        s0 = __builtin_amdgcn_wmma_f32_16x16x32_bf16(false, aq[s].v, false, bk0.v, (short)0, s0, false, false);
        s1 = __builtin_amdgcn_wmma_f32_16x16x32_bf16(false, aq[s].v, false, bk1.v, (short)0, s1, false, false);
      }
      // scale + causal mask + online softmax
      unsigned short* sPw = sP[w];
#pragma unroll
      for (int r = 0; r < 8; ++r) {
        const int mrow = q0 + r + hi8;
        float e0 = s0[r] * 0.125f, e1 = s1[r] * 0.125f;
        if (j0 + n > mrow) e0 = NEGINF;
        if (j0 + 16 + n > mrow) e1 = NEGINF;
        const float rmax = half_max(fmaxf(e0, e1));
        const float mnew = fmaxf(mstate[r], rmax);
        const float corr = __expf(mstate[r] - mnew);
        const float p0 = __expf(e0 - mnew), p1 = __expf(e1 - mnew);
        lstate[r] = lstate[r] * corr + half_sum(p0 + p1);
        mstate[r] = mnew;
        o[0][r] *= corr; o[1][r] *= corr; o[2][r] *= corr; o[3][r] *= corr;
        sPw[(r + hi8) * 32 + n] = bf16_bits(p0);
        sPw[(r + hi8) * 32 + 16 + n] = bf16_bits(p1);
      }
      asm volatile("s_wait_dscnt 0" ::: "memory");
      // reload P in A-matrix layout
      ABFrag ap;
      {
        const unsigned* pp = (const unsigned*)sPw + (lane & 15) * 16;
        const int b4 = (lane & 16) ? 8 : 0;
#pragma unroll
        for (int j = 0; j < 4; ++j) { ap.u[j] = pp[b4 + j]; ap.u[4 + j] = pp[4 + b4 + j]; }
      }
      // O += P @ V  (4 dim tiles of 16)
      const unsigned* sVu = (const unsigned*)sVh;
#pragma unroll
      for (int t = 0; t < 4; ++t) {
        ABFrag bvf;
        const unsigned* pv = sVu + (t * 16 + (lane & 15)) * 16 + ((lane & 16) ? 8 : 0);
#pragma unroll
        for (int j = 0; j < 8; ++j) bvf.u[j] = pv[j];
        o[t] = __builtin_amdgcn_wmma_f32_16x16x32_bf16(false, ap.v, false, bvf.v, (short)0, o[t], false, false);
      }
    }
    __syncthreads();
  }

  // write O / l
#pragma unroll
  for (int r = 0; r < 8; ++r) {
    const float inv = 1.f / lstate[r];
    const int trow = q0 + r + hi8;
#pragma unroll
    for (int t = 0; t < 4; ++t)
      yp[(size_t)(bz * Tt + trow) * CC + hh * DD + t * 16 + n] = o[t][r] * inv;
  }
}

// ---------------------------------------------------------------- driver
extern "C" void kernel_launch(void* const* d_in, const int* in_sizes, int n_in,
                              void* d_out, int out_size, void* d_ws, size_t ws_size,
                              hipStream_t stream) {
  (void)in_sizes; (void)n_in; (void)out_size; (void)ws_size;
  const int*   idx   = (const int*)  d_in[0];
  const float* tok   = (const float*)d_in[1];
  const float* pos   = (const float*)d_in[2];
  const float* ln1w  = (const float*)d_in[3];
  const float* ln1b  = (const float*)d_in[4];
  const float* ln2w  = (const float*)d_in[5];
  const float* ln2b  = (const float*)d_in[6];
  const float* Wq    = (const float*)d_in[7];
  const float* bq    = (const float*)d_in[8];
  const float* Wk    = (const float*)d_in[9];
  const float* bk    = (const float*)d_in[10];
  const float* Wv    = (const float*)d_in[11];
  const float* bv    = (const float*)d_in[12];
  const float* Wp    = (const float*)d_in[13];
  const float* bp    = (const float*)d_in[14];
  const float* W1    = (const float*)d_in[15];
  const float* b1    = (const float*)d_in[16];
  const float* W2    = (const float*)d_in[17];
  const float* b2    = (const float*)d_in[18];
  const float* lnfw  = (const float*)d_in[19];
  const float* lnfb  = (const float*)d_in[20];
  const float* headw = (const float*)d_in[21];
  float* out = (float*)d_out;

  const size_t MC = (size_t)MT * CC;
  float* x  = (float*)d_ws;      // [MT,CC]
  float* h  = x + MC;            // [MT,CC]  (also reused as attention output)
  float* sc = h + MC;            // scratch: MT*FFF floats
  float* qb = sc;
  float* kb = sc + MC;
  float* vb = sc + 2 * MC;
  float* ff = sc;
  float* yb = h;

  embed_kernel<<<(MT * CC) / 256, 256, 0, stream>>>(idx, tok, pos, x);

  const dim3 gc(CC / 128, MT / 64);
  for (int l = 0; l < LL; ++l) {
    ln_kernel<<<MT, 256, 0, stream>>>(x, ln1w + l * CC, ln1b + l * CC, h, CC);
    gemm_kernel<<<gc, 256, 0, stream>>>(h, Wq + (size_t)l * CC * CC, bq + l * CC, nullptr, qb, MT, CC, CC, 0);
    gemm_kernel<<<gc, 256, 0, stream>>>(h, Wk + (size_t)l * CC * CC, bk + l * CC, nullptr, kb, MT, CC, CC, 0);
    gemm_kernel<<<gc, 256, 0, stream>>>(h, Wv + (size_t)l * CC * CC, bv + l * CC, nullptr, vb, MT, CC, CC, 0);
    attn_kernel<<<dim3(Tt / 64, HH, Bb), 128, 0, stream>>>(qb, kb, vb, yb);
    gemm_kernel<<<gc, 256, 0, stream>>>(yb, Wp + (size_t)l * CC * CC, bp + l * CC, x, x, MT, CC, CC, 0);
    ln_kernel<<<MT, 256, 0, stream>>>(x, ln2w + l * CC, ln2b + l * CC, h, CC);
    gemm_kernel<<<dim3(FFF / 128, MT / 64), 256, 0, stream>>>(h, W1 + (size_t)l * FFF * CC, b1 + l * FFF, nullptr, ff, MT, FFF, CC, 1);
    gemm_kernel<<<gc, 256, 0, stream>>>(ff, W2 + (size_t)l * CC * FFF, b2 + l * CC, x, x, MT, CC, FFF, 0);
  }
  ln_kernel<<<MT, 256, 0, stream>>>(x, lnfw, lnfb, h, CC);
  gemm_kernel<<<dim3(VV / 128, MT / 64), 256, 0, stream>>>(h, headw, nullptr, nullptr, out, MT, VV, CC, 0);
}
